// Transducer_21801253995189
// MI455X (gfx1250) — matrix operand order repaired and checked by reading, not compiled, not run on hardware
//
#include <hip/hip_runtime.h>
#include <hip/hip_bf16.h>
#include <math.h>

#define Bd   8
#define Td   400
#define Ud   48
#define U1   49
#define INd  240
#define Hd   256
#define Vd   128
#define Ed   128

typedef __attribute__((ext_vector_type(16))) _Float16 v16h;
typedef __attribute__((ext_vector_type(8)))  _Float16 v8h;
typedef __attribute__((ext_vector_type(8)))  float    v8f;
typedef __attribute__((ext_vector_type(4)))  int      v4i;

#if defined(__gfx1250__) && __has_builtin(__builtin_amdgcn_global_load_async_to_lds_b128)
#define USE_ASYNC_LDS 1
#else
#define USE_ASYNC_LDS 0
#endif

// ---------------------------------------------------------------------------
// f32 -> f16 conversion with zero padding: out[Mpad x Kpad] from in[M x K(ldi)]
// ---------------------------------------------------------------------------
__global__ __launch_bounds__(256) void cvt_f16_pad(
    const float* __restrict__ in, _Float16* __restrict__ out,
    int M, int K, int ldi, int Kpad)
{
    const int idx = blockIdx.x * 256 + threadIdx.x;   // over Mpad*Kpad
    const int row = idx / Kpad;
    const int k   = idx - row * Kpad;
    float v = (row < M && k < K) ? in[(size_t)row * ldi + k] : 0.0f;
    out[idx] = (_Float16)v;
}

// ---------------------------------------------------------------------------
// GEMM: C[M,N] = A16[Mpad,Kpad] * W16[N,Kpad]^T (+ bias[N]).
// Block = 256 threads (8 waves). Block tile = 128 rows x 32 cols:
//   - the 32-col B panel (contiguous slab of W, 32*Kpad halves) is staged
//     into LDS once per block (async-to-LDS when available),
//   - wave w owns M-tile m0 = blockIdx.y*128 + w*16 and runs two
//     v_wmma_f32_16x16x32_f16 accumulators (two N-tiles), reusing each
//     A fragment twice; B fragments come from ds_load_b128.
// Fragment layouts per CDNA5 ISA §7.12.2:
//   A elem e<8 : K = k0 + hi*8 + e          (contiguous 8)
//   A elem e>=8: K = k0 + 16 + hi*8 + (e-8) (contiguous 8)
//   B elem e   : same K mapping as A
// ---------------------------------------------------------------------------
__global__ __launch_bounds__(256) void gemm_wmma_f16(
    const _Float16* __restrict__ A, const _Float16* __restrict__ W,
    const float* __restrict__ bias, float* __restrict__ C,
    int M, int Mpad, int Kpad, int ldc)
{
    const int n0   = blockIdx.x * 32;
    const int tid  = threadIdx.x;
    const int lane = tid & 31;
    const int wave = tid >> 5;

    __shared__ __align__(16) _Float16 Wlds[32 * 256];   // up to 16 KB

    // ---- stage the contiguous 32-row B panel into LDS ----
    {
        const int nh = 32 * Kpad;                       // halves in panel
        const _Float16* Wsrc = W + (size_t)n0 * Kpad;
        for (int off = tid * 8; off < nh; off += 256 * 8) {
#if USE_ASYNC_LDS
            __builtin_amdgcn_global_load_async_to_lds_b128(
                (__attribute__((address_space(1))) v4i*)(v4i*)(void*)(Wsrc + off),
                (__attribute__((address_space(3))) v4i*)(v4i*)(void*)(&Wlds[off]),
                0, 0);
#else
            *(v8h*)(&Wlds[off]) = *(const v8h*)(Wsrc + off);
#endif
        }
#if USE_ASYNC_LDS
#if __has_builtin(__builtin_amdgcn_s_wait_asynccnt)
        __builtin_amdgcn_s_wait_asynccnt(0);
#else
        asm volatile("s_wait_asynccnt 0x0" ::: "memory");
#endif
#endif
    }
    __syncthreads();

    const int m0 = blockIdx.y * 128 + wave * 16;
    if (m0 >= Mpad) return;                              // whole-wave skip

    const int l15 = lane & 15;
    const int hi  = lane >> 4;
    const _Float16* Arow = A + (size_t)(m0 + l15) * Kpad;   // rows padded
    const _Float16* W0   = &Wlds[(size_t)l15 * Kpad];        // N-tile 0
    const _Float16* W1   = &Wlds[(size_t)(16 + l15) * Kpad]; // N-tile 1

    v8f acc0 = {}, acc1 = {};
    for (int k0 = 0; k0 < Kpad; k0 += 32) {
        v8h a0 = *(const v8h*)(Arow + k0 + hi * 8);
        v8h a1 = *(const v8h*)(Arow + k0 + 16 + hi * 8);
        v16h a = __builtin_shufflevector(a0, a1, 0,1,2,3,4,5,6,7,8,9,10,11,12,13,14,15);

        v8h b00 = *(const v8h*)(W0 + k0 + hi * 8);
        v8h b01 = *(const v8h*)(W0 + k0 + 16 + hi * 8);
        v16h b0v = __builtin_shufflevector(b00, b01, 0,1,2,3,4,5,6,7,8,9,10,11,12,13,14,15);
        acc0 = __builtin_amdgcn_wmma_f32_16x16x32_f16(
            false, a, false, b0v, (short)0, acc0, false, false);

        v8h b10 = *(const v8h*)(W1 + k0 + hi * 8);
        v8h b11 = *(const v8h*)(W1 + k0 + 16 + hi * 8);
        v16h b1v = __builtin_shufflevector(b10, b11, 0,1,2,3,4,5,6,7,8,9,10,11,12,13,14,15);
        acc1 = __builtin_amdgcn_wmma_f32_16x16x32_f16(
            false, a, false, b1v, (short)0, acc1, false, false);
    }

    const int   col0 = n0 + l15;
    const int   col1 = n0 + 16 + l15;
    const int   mb   = hi ? 8 : 0;
    const float bv0  = bias ? bias[col0] : 0.0f;
    const float bv1  = bias ? bias[col1] : 0.0f;
    #pragma unroll
    for (int r = 0; r < 8; ++r) {
        int rr = m0 + mb + r;
        if (rr < M) {
            C[(size_t)rr * ldc + col0] = acc0[r] + bv0;
            C[(size_t)rr * ldc + col1] = acc1[r] + bv1;
        }
    }
}

// ---------------------------------------------------------------------------
// float4 dot helper: 256-wide dot, b128 loads for both operands.
// ---------------------------------------------------------------------------
__device__ __forceinline__ float dot256(const float* __restrict__ w,
                                        const float* __restrict__ h)
{
    float s = 0.0f;
    #pragma unroll 8
    for (int k = 0; k < Hd; k += 4) {
        float4 wv = *(const float4*)(w + k);
        float4 hv = *(const float4*)(h + k);
        s += wv.x * hv.x + wv.y * hv.y + wv.z * hv.z + wv.w * hv.w;
    }
    return s;
}

// ---------------------------------------------------------------------------
// GRU recurrence. One block per batch, 256 threads = hidden units, h in LDS.
// ---------------------------------------------------------------------------
__global__ __launch_bounds__(256) void gru_kernel(
    const float* __restrict__ xg, const float* __restrict__ Whh,
    const float* __restrict__ bhh, float* __restrict__ hout, int T)
{
    const int b = blockIdx.x;
    const int i = threadIdx.x;
    __shared__ __align__(16) float h[Hd];
    h[i] = 0.0f;
    __syncthreads();

    const float* wr = Whh + (size_t)(i)        * Hd;
    const float* wz = Whh + (size_t)(Hd + i)   * Hd;
    const float* wn = Whh + (size_t)(2*Hd + i) * Hd;
    const float br = bhh[i], bz = bhh[Hd + i], bn = bhh[2*Hd + i];

    for (int t = 0; t < T; ++t) {
        const float* xrow = xg + ((size_t)(b * T + t)) * (3 * Hd);
        float hr = br + dot256(wr, h);
        float hz = bz + dot256(wz, h);
        float hn = bn + dot256(wn, h);
        float r = 1.0f / (1.0f + expf(-(xrow[i] + hr)));
        float z = 1.0f / (1.0f + expf(-(xrow[Hd + i] + hz)));
        float n = tanhf(xrow[2*Hd + i] + r * hn);
        float hnew = (1.0f - z) * n + z * h[i];
        __syncthreads();
        h[i] = hnew;
        __syncthreads();
        hout[((size_t)(b * T + t)) * Hd + i] = hnew;
    }
}

// ---------------------------------------------------------------------------
// LSTM recurrence (decoder).
// ---------------------------------------------------------------------------
__global__ __launch_bounds__(256) void lstm_kernel(
    const float* __restrict__ xg, const float* __restrict__ Whh,
    const float* __restrict__ bhh, float* __restrict__ hout, int T)
{
    const int b = blockIdx.x;
    const int i = threadIdx.x;
    __shared__ __align__(16) float h[Hd];
    float c = 0.0f;
    h[i] = 0.0f;
    __syncthreads();

    const float* wi = Whh + (size_t)(i)        * Hd;
    const float* wf = Whh + (size_t)(Hd + i)   * Hd;
    const float* wg = Whh + (size_t)(2*Hd + i) * Hd;
    const float* wo = Whh + (size_t)(3*Hd + i) * Hd;
    const float bi = bhh[i], bf_ = bhh[Hd+i], bg = bhh[2*Hd+i], bo = bhh[3*Hd+i];

    for (int t = 0; t < T; ++t) {
        const float* xrow = xg + ((size_t)(b * T + t)) * (4 * Hd);
        float gi = bi  + xrow[i]        + dot256(wi, h);
        float gf = bf_ + xrow[Hd + i]   + dot256(wf, h);
        float gg = bg  + xrow[2*Hd + i] + dot256(wg, h);
        float go = bo  + xrow[3*Hd + i] + dot256(wo, h);
        c = (1.0f / (1.0f + expf(-gf))) * c
          + (1.0f / (1.0f + expf(-gi))) * tanhf(gg);
        float hn = (1.0f / (1.0f + expf(-go))) * tanhf(c);
        __syncthreads();
        h[i] = hn;
        __syncthreads();
        hout[((size_t)(b * T + t)) * Hd + i] = hn;
    }
}

// ---------------------------------------------------------------------------
// Embedding lookup with prepended blank and embed[BLANK] forced to 0.
// ---------------------------------------------------------------------------
__global__ __launch_bounds__(128) void embed_kernel(
    const int* __restrict__ ys, const float* __restrict__ embed,
    float* __restrict__ out)
{
    const int bu = blockIdx.x;            // b*U1 + u
    const int b  = bu / U1, u = bu % U1;
    const int e  = threadIdx.x;
    const int tok = (u == 0) ? 0 : ys[b * Ud + (u - 1)];
    out[(size_t)bu * Ed + e] = (tok == 0) ? 0.0f : embed[(size_t)tok * Ed + e];
}

// ---------------------------------------------------------------------------
// Joint: tile of 16 t's at fixed (b,u):
//   hj = tanh(fW + gW + fc1_b) -> LDS (f16, 16B-aligned rows)
//   logits = hj @ fc2W16^T + fc2_b via WMMA (A frags from ds_load_b128)
//   emit blank/label log-probs only.
// ---------------------------------------------------------------------------
#define HJ_LD (Hd + 8)   // 264 halves = 528B rows (16B multiple)

__global__ __launch_bounds__(128) void joint_kernel(
    const float* __restrict__ fW, const float* __restrict__ gW,
    const float* __restrict__ fc1_b, const _Float16* __restrict__ fc2W16,
    const float* __restrict__ fc2_b, const int* __restrict__ ys,
    float* __restrict__ blank_lp, float* __restrict__ lab_lp)
{
    const int t0   = blockIdx.x * 16;
    const int u    = blockIdx.y;
    const int b    = blockIdx.z;
    const int tid  = threadIdx.x;
    const int lane = tid & 31;
    const int wave = tid >> 5;

    __shared__ __align__(16) _Float16 hj[16][HJ_LD];
    __shared__ __align__(16) float    logits[16][Vd];
    __shared__ float lse_s[16];

    const float* gRow = gW + ((size_t)b * U1 + u) * Hd;
    for (int idx = tid; idx < 16 * Hd; idx += 128) {
        int m = idx >> 8, hh = idx & (Hd - 1);
        float v = fW[((size_t)(b * Td + t0 + m)) * Hd + hh] + gRow[hh] + fc1_b[hh];
        hj[m][hh] = (_Float16)tanhf(v);
    }
    __syncthreads();

    const int l15 = lane & 15;
    const int hi  = lane >> 4;
    const _Float16* hrow = &hj[l15][0];
    for (int nt = wave; nt < 8; nt += 4) {
        const int n0  = nt * 16;
        const int col = n0 + l15;
        const _Float16* Wrow = fc2W16 + (size_t)col * Hd;
        v8f acc = {};
        for (int k0 = 0; k0 < Hd; k0 += 32) {
            v8h a0 = *(const v8h*)(hrow + k0 + hi * 8);
            v8h a1 = *(const v8h*)(hrow + k0 + 16 + hi * 8);
            v8h b0 = *(const v8h*)(Wrow + k0 + hi * 8);
            v8h b1 = *(const v8h*)(Wrow + k0 + 16 + hi * 8);
            v16h a = __builtin_shufflevector(a0, a1, 0,1,2,3,4,5,6,7,8,9,10,11,12,13,14,15);
            v16h bb = __builtin_shufflevector(b0, b1, 0,1,2,3,4,5,6,7,8,9,10,11,12,13,14,15);
            acc = __builtin_amdgcn_wmma_f32_16x16x32_f16(
                false, a, false, bb, (short)0, acc, false, false);
        }
        const int mb = hi ? 8 : 0;
        const float bv = fc2_b[col];
        #pragma unroll
        for (int r = 0; r < 8; ++r)
            logits[mb + r][col] = acc[r] + bv;
    }
    __syncthreads();

    if (tid < 16) {
        float mx = -1e30f;
        for (int v = 0; v < Vd; ++v) mx = fmaxf(mx, logits[tid][v]);
        float s = 0.0f;
        for (int v = 0; v < Vd; ++v) s += expf(logits[tid][v] - mx);
        lse_s[tid] = mx + logf(s);
    }
    __syncthreads();

    if (tid < 16) {
        const int t = t0 + tid;
        const float lse = lse_s[tid];
        blank_lp[((size_t)(b * Td + t)) * U1 + u] = logits[tid][0] - lse;
        if (u < Ud) {
            int lab = ys[b * Ud + u];
            lab_lp[((size_t)(b * Td + t)) * Ud + u] = logits[tid][lab] - lse;
        }
    }
}

// ---------------------------------------------------------------------------
// RNN-T alpha recursion + mean NLL.
// ---------------------------------------------------------------------------
__device__ __forceinline__ float logaddexpf_(float a, float b) {
    float m = fmaxf(a, b);
    float d = fminf(a, b) - m;
    return m + log1pf(expf(d));
}

__global__ __launch_bounds__(32) void loss_kernel(
    const int* __restrict__ xlen, const int* __restrict__ ylen,
    const float* __restrict__ blank_lp, const float* __restrict__ lab_lp,
    float* __restrict__ out)
{
    __shared__ float lls[Bd];
    const int b = threadIdx.x;
    if (b < Bd) {
        const float* bl = blank_lp + (size_t)b * Td * U1;
        const float* la = lab_lp   + (size_t)b * Td * Ud;
        float alpha[U1];
        alpha[0] = 0.0f;
        for (int u = 1; u < U1; ++u) alpha[u] = alpha[u - 1] + la[u - 1];
        const int xl = xlen[b] - 1;
        const int yl = ylen[b];
        float ll = 0.0f;
        if (xl == 0) ll = alpha[yl] + bl[yl];
        for (int t = 1; t < Td; ++t) {
            float carry = alpha[0] + bl[(size_t)(t - 1) * U1 + 0];
            alpha[0] = carry;
            for (int u = 1; u < U1; ++u) {
                float up = alpha[u] + bl[(size_t)(t - 1) * U1 + u];
                carry = logaddexpf_(up, carry + la[(size_t)t * Ud + (u - 1)]);
                alpha[u] = carry;
            }
            if (t == xl) ll = alpha[yl] + bl[(size_t)t * U1 + yl];
        }
        lls[b] = ll;
    }
    __syncthreads();
    if (threadIdx.x == 0) {
        float s = 0.0f;
        for (int i = 0; i < Bd; ++i) s += lls[i];
        out[0] = -s / (float)Bd;
    }
}

// ---------------------------------------------------------------------------
extern "C" void kernel_launch(void* const* d_in, const int* in_sizes, int n_in,
                              void* d_out, int out_size, void* d_ws, size_t ws_size,
                              hipStream_t stream) {
    (void)in_sizes; (void)n_in; (void)out_size; (void)ws_size;

    const float* xs    = (const float*)d_in[0];
    const int*   ys    = (const int*)  d_in[1];
    const int*   xlen  = (const int*)  d_in[2];
    const int*   ylen  = (const int*)  d_in[3];
    const float* embed = (const float*)d_in[4];
    const float* eWih0 = (const float*)d_in[5];
    const float* eWhh0 = (const float*)d_in[6];
    const float* ebih0 = (const float*)d_in[7];
    const float* ebhh0 = (const float*)d_in[8];
    const float* eWih1 = (const float*)d_in[9];
    const float* eWhh1 = (const float*)d_in[10];
    const float* ebih1 = (const float*)d_in[11];
    const float* ebhh1 = (const float*)d_in[12];
    const float* linW  = (const float*)d_in[13];
    const float* linB  = (const float*)d_in[14];
    const float* dWih  = (const float*)d_in[15];
    const float* dWhh  = (const float*)d_in[16];
    const float* dbih  = (const float*)d_in[17];
    const float* dbhh  = (const float*)d_in[18];
    const float* fc1W  = (const float*)d_in[19];
    const float* fc1b  = (const float*)d_in[20];
    const float* fc2W  = (const float*)d_in[21];
    const float* fc2b  = (const float*)d_in[22];

    float* ws = (float*)d_ws;
    // f32 regions (floats):
    float* bufA     = ws;                        // 2,457,600  (xg0/xg1/dec_xg)
    float* bufB     = bufA     + 2457600;        //   819,200  (h0, then f)
    float* bufC     = bufB     + 819200;         //   819,200  (h1, then fW)
    float* bufEmb   = bufC     + 819200;         //    50,176
    float* bufG     = bufEmb   + 50176;          //   100,352
    float* bufGW    = bufG     + 100352;         //   100,352
    float* bufBlank = bufGW    + 100352;         //   156,800
    float* bufLab   = bufBlank + 156800;         //   153,600
    // f16 regions (halves):
    _Float16* A16   = (_Float16*)(bufLab + 153600);  // 819,200 halves
    _Float16* W16   = A16 + 819200;                  // 262,144 halves
    _Float16* F2W16 = W16 + 262144;                  //  32,768 halves

    const int BT  = Bd * Td;     // 3200
    const int BU  = Bd * U1;     // 392
    const int BUp = 400;         // padded rows for decoder-side GEMMs

    #define CVT(in, out, M, K, ldi, Mpad, Kpad) \
        cvt_f16_pad<<<((Mpad)*(Kpad))/256, 256, 0, stream>>>(in, out, M, K, ldi, Kpad)
    #define GEMM(A, W, bias, C, M, Mpad, Kpad, N) \
        gemm_wmma_f16<<<dim3((N)/32, ((Mpad)+127)/128), 256, 0, stream>>>( \
            A, W, bias, C, M, Mpad, Kpad, N)

    // ---- Encoder layer 0: xg0 = xs @ Wih0^T + bih0
    CVT(xs,    A16, BT,   INd, INd, BT,   256);
    CVT(eWih0, W16, 3*Hd, INd, INd, 3*Hd, 256);
    GEMM(A16, W16, ebih0, bufA, BT, BT, 256, 3*Hd);
    gru_kernel<<<Bd, Hd, 0, stream>>>(bufA, eWhh0, ebhh0, bufB, Td);

    // ---- Encoder layer 1
    CVT(bufB,  A16, BT,   Hd, Hd, BT,   256);
    CVT(eWih1, W16, 3*Hd, Hd, Hd, 3*Hd, 256);
    GEMM(A16, W16, ebih1, bufA, BT, BT, 256, 3*Hd);
    gru_kernel<<<Bd, Hd, 0, stream>>>(bufA, eWhh1, ebhh1, bufC, Td);

    // ---- Encoder linear: f = h1 @ linW^T + linB -> bufB
    CVT(bufC, A16, BT, Hd, Hd, BT, 256);
    CVT(linW, W16, Hd, Hd, Hd, Hd, 256);
    GEMM(A16, W16, linB, bufB, BT, BT, 256, Hd);

    // ---- fW = f @ Wf^T (Wf = fc1_W[:, :H], row stride 2H) -> bufC
    CVT(bufB, A16, BT, Hd, Hd,   BT, 256);
    CVT(fc1W, W16, Hd, Hd, 2*Hd, Hd, 256);
    GEMM(A16, W16, nullptr, bufC, BT, BT, 256, Hd);

    // ---- Decoder
    embed_kernel<<<BU, Ed, 0, stream>>>(ys, embed, bufEmb);
    CVT(bufEmb, A16, BU,   Ed, Ed, BUp,  128);
    CVT(dWih,   W16, 4*Hd, Ed, Ed, 4*Hd, 128);
    GEMM(A16, W16, dbih, bufA, BU, BUp, 128, 4*Hd);
    lstm_kernel<<<Bd, Hd, 0, stream>>>(bufA, dWhh, dbhh, bufG, U1);

    // ---- gW = g @ Wg^T (Wg = fc1_W[:, H:], row stride 2H) -> bufGW
    CVT(bufG,      A16, BU, Hd, Hd,   BUp, 256);
    CVT(fc1W + Hd, W16, Hd, Hd, 2*Hd, Hd,  256);
    GEMM(A16, W16, nullptr, bufGW, BU, BUp, 256, Hd);

    // ---- Joint + loss
    CVT(fc2W, F2W16, Vd, Hd, Hd, Vd, 256);
    joint_kernel<<<dim3(Td/16, U1, Bd), 128, 0, stream>>>(
        bufC, bufGW, fc1b, F2W16, fc2b, ys, bufBlank, bufLab);
    loss_kernel<<<1, 32, 0, stream>>>(xlen, ylen, bufBlank, bufLab, (float*)d_out);

    #undef CVT
    #undef GEMM
}
